// LiquidStateMachine_68126771249181
// MI455X (gfx1250) — matrix operand order repaired and checked by reading, compiled, not run on hardware
//
#include <hip/hip_runtime.h>

// ---------------------------------------------------------------------------
// LSM: B=128, T=2048, IN=128, LIQ=512, readout 512->256->128->64
// Batch rows independent -> 8 WGs, one 16-row batch tile each, no inter-WG
// sync. All GEMMs via v_wmma_f32_16x16x32_bf16, f32 accumulate.
// v6: v5's in-place double-buffered B prefetch, but with the k-loop forced
//     rolled (#pragma unroll 1) so the compiler can neither fully unroll nor
//     hoist the whole weight set across the t-loop (v5's scratch-spill bug).
// ---------------------------------------------------------------------------

typedef __attribute__((ext_vector_type(16))) __bf16 v16bf;
typedef __attribute__((ext_vector_type(8)))  float  v8f;

#define LSM_B   128
#define LSM_T   2048
#define LSM_IN  128
#define LSM_LIQ 512
#define LSM_K   (LSM_IN + LSM_LIQ)      // 640
#define LSM_H1  256
#define LSM_H2  128
#define LSM_OUT 64

// workspace layout (dword offsets)
#define OFF_WB  0                        // 20 kb * 32 nt * 32 lanes * 8 dw = 163840
#define OFF_W1  163840                   // 16 * 16 * 256 = 65536
#define OFF_W2  229376                   // 8 * 8 * 256   = 16384
#define OFF_W3  245760                   // 4 * 4 * 256   = 4096
#define OFF_BP  249856                   // 512 f32 (folded bias)

__device__ __forceinline__ unsigned short f2bf(float f) {
  union { float f; unsigned u; } c; c.f = f;
  unsigned u = c.u + 0x7FFFu + ((c.u >> 16) & 1u);   // round-to-nearest-even
  return (unsigned short)(u >> 16);
}

// gfx1250 hardware V_TANH_F32 (probe-confirmed: lowers to v_tanh_f32_e32).
__device__ __forceinline__ float fast_tanh(float x) {
#if __has_builtin(__builtin_amdgcn_tanhf)
  return __builtin_amdgcn_tanhf(x);
#elif __has_builtin(__builtin_amdgcn_tanh_f32)
  return __builtin_amdgcn_tanh_f32(x);
#else
  float e = __builtin_amdgcn_exp2f(x * 2.885390081777927f);  // e^(2x)
  return 1.0f - 2.0f * __builtin_amdgcn_rcpf(e + 1.0f);
#endif
}

// B-matrix 32x16 bf16 fragment K offset: lanes 0-15 K in {0..7,16..23}, lanes16-31 {8..15,24..31}
__device__ __forceinline__ int frag_bk0(int lane, int v) {
  return 2 * v + (v >= 4 ? 8 : 0) + (((lane >> 4) & 1) ? 8 : 0);
}

union FragU { uint4 u[2]; v16bf v; };

// Load A fragment (16 rows x 32 K of bf16) from LDS, row-major, 16B-aligned rows.
// A layout (ISA 7.12.2): lanes 0-15 K in {0..7,16..23}, lanes 16-31 {8..15,24..31}
// -> two contiguous uint4 (ds_load_b128) per fragment.
__device__ __forceinline__ v16bf load_afrag(const unsigned short* base, int stride,
                                            int kstart, int lane) {
  int m    = lane & 15;
  int half = (lane >> 4) & 1;
  const unsigned short* p = base + m * stride + kstart;
  FragU f;
  f.u[0] = *(const uint4*)(p + (half ? 8  : 0));    // VGPRs 0-3
  f.u[1] = *(const uint4*)(p + (half ? 24 : 16));   // VGPRs 4-7
  return f.v;
}

// Load pre-packed B fragment (32 K x 16 N bf16) from global: 8 contiguous dwords/lane.
__device__ __forceinline__ v16bf load_bfrag(const unsigned* __restrict__ wp,
                                            int tile, int lane) {
  const uint4* p = (const uint4*)(wp + ((size_t)tile * 32 + lane) * 8);
  FragU f;
  f.u[0] = p[0];
  f.u[1] = p[1];
  return f.v;
}

__device__ __forceinline__ int imin(int a, int b) { return a < b ? a : b; }

// One M16xN16 output tile over KB k-blocks (K = KB*32). Packed-weight tile
// index = kb*NT + wave. Period-2 body with in-place B reloads: each buffer is
// overwritten right after its consuming WMMA (distance-2 prefetch, no movs).
// #pragma unroll 1 keeps the loop rolled: prevents full unroll + cross-t
// hoisting of the entire weight set (which spilled to scratch in v5).
template <int KB, int NT>
__device__ __forceinline__ v8f gemm_tile(const unsigned short* abase, int astride,
                                         const unsigned* __restrict__ wp,
                                         int wave, int lane, float bias) {
  v8f acc;
  #pragma unroll
  for (int i = 0; i < 8; ++i) acc[i] = bias;
  v16bf b0 = load_bfrag(wp, 0 * NT + wave, lane);
  v16bf b1 = load_bfrag(wp, 1 * NT + wave, lane);
  #pragma unroll 1
  for (int kb = 0; kb < KB; kb += 2) {
    v16bf a0 = load_afrag(abase, astride, kb * 32, lane);
    acc = __builtin_amdgcn_wmma_f32_16x16x32_bf16(false, a0, false, b0,
                                                  (short)0, acc, false, false);
    b0 = load_bfrag(wp, imin(kb + 2, KB - 1) * NT + wave, lane);
    v16bf a1 = load_afrag(abase, astride, (kb + 1) * 32, lane);
    acc = __builtin_amdgcn_wmma_f32_16x16x32_bf16(false, a1, false, b1,
                                                  (short)0, acc, false, false);
    b1 = load_bfrag(wp, imin(kb + 3, KB - 1) * NT + wave, lane);
  }
  return acc;
}

// ---------------------------------------------------------------------------
// Prep: fold BN into weights/bias, pack all weight matrices into B-fragment
// order as bf16 pairs. blocks 0..975 pack one 16x16-output (32-K) tile each;
// block 976 computes the folded liquid bias.
// ---------------------------------------------------------------------------
__global__ void lsm_prep(const float* __restrict__ W,  const float* __restrict__ b,
                         const float* __restrict__ gamma, const float* __restrict__ beta,
                         const float* __restrict__ mmean, const float* __restrict__ mvar,
                         const float* __restrict__ W1, const float* __restrict__ W2,
                         const float* __restrict__ W3, unsigned* __restrict__ ws) {
  int blk = blockIdx.x;
  int t   = threadIdx.x;

  if (blk == 976) {                       // folded bias: b' = b*scale + shift
    for (int j = t; j < LSM_LIQ; j += 256) {
      float sc = gamma[j] * rsqrtf(mvar[j] + 1e-3f);
      float sh = beta[j] - mmean[j] * sc;
      ((float*)(ws + OFF_BP))[j] = b[j] * sc + sh;
    }
    return;
  }

  const float* src; int NT, N, off, ltid; bool scl = false;
  if (blk < 640)      { src = W;  NT = 32; N = 512; off = OFF_WB; ltid = blk;       scl = true; }
  else if (blk < 896) { src = W1; NT = 16; N = 256; off = OFF_W1; ltid = blk - 640; }
  else if (blk < 960) { src = W2; NT = 8;  N = 128; off = OFF_W2; ltid = blk - 896; }
  else                { src = W3; NT = 4;  N = 64;  off = OFF_W3; ltid = blk - 960; }

  int lane = t >> 3, v = t & 7;
  int kb = ltid / NT, nt = ltid % NT;
  int n  = nt * 16 + (lane & 15);
  float cs = 1.0f;
  if (scl) cs = gamma[n] * rsqrtf(mvar[n] + 1e-3f);   // column scale from BN fold
  int kk = kb * 32 + frag_bk0(lane, v);
  float v0 = src[(size_t)kk * N + n] * cs;
  float v1 = src[(size_t)(kk + 1) * N + n] * cs;
  ws[(size_t)off + ((size_t)ltid * 32 + lane) * 8 + v] =
      (unsigned)f2bf(v0) | ((unsigned)f2bf(v1) << 16);
}

// ---------------------------------------------------------------------------
// Main persistent kernel: 8 blocks (one per 16-row batch tile) x 1024 threads
// (32 wave32s). Liquid state lives in per-wave registers (wave w owns liquid
// cols [16w,16w+16)), bf16 shadow in LDS 'combined[:,128:]' for A-fragments.
// ---------------------------------------------------------------------------
__global__ __launch_bounds__(1024) void lsm_main(
    const float* __restrict__ X, const unsigned* __restrict__ ws,
    const float* __restrict__ b1, const float* __restrict__ b2,
    const float* __restrict__ b3, float* __restrict__ out) {
  __shared__ __align__(16) unsigned short combined[16][LSM_K];   // [x_t | state] bf16
  __shared__ __align__(16) unsigned short h1s[16][LSM_H1];
  __shared__ __align__(16) unsigned short h2s[16][LSM_H2];

  const unsigned* WbP = ws + OFF_WB;
  const unsigned* W1P = ws + OFF_W1;
  const unsigned* W2P = ws + OFF_W2;
  const unsigned* W3P = ws + OFF_W3;
  const float* bprime = (const float*)(ws + OFF_BP);

  const int tid  = threadIdx.x;
  const int wave = tid >> 5;
  const int lane = tid & 31;
  const int mrow = lane & 15;
  const int half = (lane >> 4) & 1;
  const int r0   = blockIdx.x * 16;          // batch rows [r0, r0+16)

  // this thread's two x-tile elements (16*128 elems / 1024 threads)
  const int xm0 = tid >> 7,          xk0 = tid & 127;
  const int xm1 = (tid + 1024) >> 7, xk1 = (tid + 1024) & 127;
  const size_t xrow0 = ((size_t)(r0 + xm0)) * LSM_T;
  const size_t xrow1 = ((size_t)(r0 + xm1)) * LSM_T;

  float s[8];                                 // persistent liquid state (this lane's slots)
  #pragma unroll
  for (int i = 0; i < 8; ++i) s[i] = 0.0f;

  for (int idx = tid; idx < 16 * LSM_LIQ; idx += 1024)   // zero state shadow
    combined[idx >> 9][LSM_IN + (idx & 511)] = 0;

  // prologue: x(0)
  combined[xm0][xk0] = f2bf(X[(xrow0 + 0) * LSM_IN + xk0]);
  combined[xm1][xk1] = f2bf(X[(xrow1 + 0) * LSM_IN + xk1]);

  const int colL = wave * 16 + mrow;          // liquid column this lane owns
  const float bpc = bprime[colL];
  const float b1c = (wave < 16) ? b1[wave * 16 + mrow] : 0.0f;
  const float b2c = (wave <  8) ? b2[wave * 16 + mrow] : 0.0f;
  const float b3c = (wave <  4) ? b3[wave * 16 + mrow] : 0.0f;

  __syncthreads();

  for (int t = 0; t < LSM_T; ++t) {
    // ---- issue prefetch of x(t+1) early; lands while WMMAs run ----
    float nx0 = 0.0f, nx1 = 0.0f;
    if (t + 1 < LSM_T) {
      nx0 = X[(xrow0 + (t + 1)) * LSM_IN + xk0];
      nx1 = X[(xrow1 + (t + 1)) * LSM_IN + xk1];
    }

    // ---- stage 1: liquid GEMM (M16 x N512 x K640) over [x|state] ----
    {
      v8f acc = gemm_tile<20, 32>(&combined[0][0], LSM_K, WbP, wave, lane, bpc);
      // leaky tanh update in registers first ...
      #pragma unroll
      for (int i = 0; i < 8; ++i)
        s[i] = 0.9f * s[i] + 0.1f * fast_tanh(acc[i]);
    }
    __syncthreads();   // all waves done READING old state
    // ... then commit new state to the shared bf16 shadow (race-free)
    #pragma unroll
    for (int i = 0; i < 8; ++i) {
      int m = i + (half ? 8 : 0);
      combined[m][LSM_IN + colL] = f2bf(s[i]);
    }
    __syncthreads();   // new state visible

    // ---- stage 2: h1 = relu(state @ W1 + b1)  (M16 x N256 x K512) ----
    if (wave < 16) {
      v8f acc = gemm_tile<16, 16>(&combined[0][LSM_IN], LSM_K, W1P, wave, lane, b1c);
      int col = wave * 16 + mrow;
      #pragma unroll
      for (int i = 0; i < 8; ++i) {
        int m = i + (half ? 8 : 0);
        float vv = acc[i] > 0.0f ? acc[i] : 0.0f;
        h1s[m][col] = f2bf(vv);
      }
    }
    __syncthreads();

    // ---- stage 3: h2 = relu(h1 @ W2 + b2)  (M16 x N128 x K256) ----
    if (wave < 8) {
      v8f acc = gemm_tile<8, 8>(&h1s[0][0], LSM_H1, W2P, wave, lane, b2c);
      int col = wave * 16 + mrow;
      #pragma unroll
      for (int i = 0; i < 8; ++i) {
        int m = i + (half ? 8 : 0);
        float vv = acc[i] > 0.0f ? acc[i] : 0.0f;
        h2s[m][col] = f2bf(vv);
      }
    }
    __syncthreads();

    // ---- stage 4: y = h2 @ W3 + b3  (M16 x N64 x K128) -> global out ----
    if (wave < 4) {
      v8f acc = gemm_tile<4, 4>(&h2s[0][0], LSM_H2, W3P, wave, lane, b3c);
      int col = wave * 16 + mrow;
      #pragma unroll
      for (int i = 0; i < 8; ++i) {
        int m = i + (half ? 8 : 0);
        out[(((size_t)(r0 + m)) * LSM_T + t) * LSM_OUT + col] = acc[i];
      }
    }

    // ---- commit prefetched x(t+1); stage-1 reads of x(t) finished before
    //      the post-stage-1 barrier, so this is race-free ----
    if (t + 1 < LSM_T) {
      combined[xm0][xk0] = f2bf(nx0);
      combined[xm1][xk1] = f2bf(nx1);
    }
    __syncthreads();
  }

  // ---- final state: [B, LIQ] f32 appended after outputs ----
  #pragma unroll
  for (int i = 0; i < 8; ++i) {
    int m = i + (half ? 8 : 0);
    out[(size_t)LSM_B * LSM_T * LSM_OUT + (size_t)(r0 + m) * LSM_LIQ + colL] = s[i];
  }
}

// ---------------------------------------------------------------------------
extern "C" void kernel_launch(void* const* d_in, const int* in_sizes, int n_in,
                              void* d_out, int out_size, void* d_ws, size_t ws_size,
                              hipStream_t stream) {
  (void)in_sizes; (void)n_in; (void)out_size; (void)ws_size;
  const float* X     = (const float*)d_in[0];
  const float* W     = (const float*)d_in[1];
  const float* b     = (const float*)d_in[2];
  const float* gamma = (const float*)d_in[3];
  const float* beta  = (const float*)d_in[4];
  const float* mmean = (const float*)d_in[5];
  const float* mvar  = (const float*)d_in[6];
  const float* W1    = (const float*)d_in[7];
  const float* b1    = (const float*)d_in[8];
  const float* W2    = (const float*)d_in[9];
  const float* b2    = (const float*)d_in[10];
  const float* W3    = (const float*)d_in[11];
  const float* b3    = (const float*)d_in[12];
  float*    out = (float*)d_out;
  unsigned* ws  = (unsigned*)d_ws;

  lsm_prep<<<dim3(977), dim3(256), 0, stream>>>(W, b, gamma, beta, mmean, mvar,
                                                W1, W2, W3, ws);
  lsm_main<<<dim3(8), dim3(1024), 0, stream>>>(X, ws, b1, b2, b3, out);
}
